// MLPVolatilityModelBasic_74113955660469
// MI455X (gfx1250) — matrix operand order repaired and checked by reading, not compiled, not run on hardware
//
#include <hip/hip_runtime.h>
#include <hip/hip_bf16.h>
#include <math.h>

// ---------------------------------------------------------------------------
// MLP-GARCH volatility NLL for MI455X (gfx1250, wave32, WMMA + TDM).
//
// u2_lags is a Hankel matrix: with w reversed along the lag axis, row t of
// the A-operand is u2_full[t .. t+63] (contiguous). The 500000 x 64 x 256
// GEMM runs on V_WMMA_F32_16X16X32_F16. Each wave processes 32 rows per
// iteration (two 16-row tiles) so every B fragment fetched from LDS feeds
// 4 WMMAs. beta (>0) is folded into the f16 B matrix and bias (relu is
// positively homogeneous); the u2 fp16 scale (256) is folded into the bias
// and removed once per row tile; relu is a single v_med3_f32.
// B (32 KB f16, transposed so fragments are contiguous b128 LDS reads) is
// staged into LDS via the Tensor Data Mover (tensor_load_to_lds +
// s_wait_tensorcnt), with a benign-race cooperative copy as semantic hedge.
// ---------------------------------------------------------------------------

typedef __attribute__((ext_vector_type(16))) _Float16     v16h;
typedef __attribute__((ext_vector_type(8)))  _Float16     v8h;
typedef __attribute__((ext_vector_type(8)))  float        v8f;
typedef __attribute__((ext_vector_type(4)))  unsigned int v4u;
typedef __attribute__((ext_vector_type(8)))  int          v8i;
typedef __attribute__((ext_vector_type(4)))  int          v4i;

#define QLAG   64
#define NKU    256
#define U2SCALE 256.0f
#define INV_U2SCALE (1.0f/256.0f)

static __device__ __forceinline__ float softplusf(float x) {
    return (x > 20.0f) ? x : log1pf(__expf(x));
}

static __device__ __forceinline__ v16h cat8(v8h lo, v8h hi) {
    return __builtin_shufflevector(lo, hi, 0,1,2,3,4,5,6,7,8,9,10,11,12,13,14,15);
}

// relu as a single v_med3_f32 (median{x, 0, FLT_MAX} == max(x,0) for finite x)
static __device__ __forceinline__ float reluf(float x) {
    return __builtin_amdgcn_fmed3f(x, 0.0f, __FLT_MAX__);
}

// Issue a 1-row TDM copy of `ndwords` dwords from global `src` to LDS offset
// `lds_off`. Descriptor layout per CDNA5 ISA section 8 (D# groups 0/1).
static __device__ __forceinline__ void tdm_load_to_lds(const void* src,
                                                       unsigned int lds_off,
                                                       unsigned int ndwords) {
    unsigned long long ga = (unsigned long long)src;
    v4u g0 = { 1u,                                   // count=1, user desc
               lds_off,                              // lds_addr (bytes)
               (unsigned int)ga,                     // global_addr[31:0]
               (unsigned int)((ga >> 32) & 0x01FFFFFFu) | (2u << 30) }; // type=2
    v8i g1 = { (int)(2u << 16),                      // data_size=4B
               (int)(ndwords << 16),                 // tensor_dim0[15:0] @63:48
               (int)(1u << 16),                      // tensor_dim1 = 1
               (int)(ndwords << 16),                 // tile_dim0 @127:112
               1,                                    // tile_dim1 = 1
               (int)ndwords,                         // tensor_dim0_stride lo
               0, 0 };
    v4i g2 = { 0, 0, 0, 0 };
    v4i g3 = { 0, 0, 0, 0 };
#if defined(__clang_major__) && (__clang_major__ >= 23)
    v8i g4 = { 0, 0, 0, 0, 0, 0, 0, 0 };
    __builtin_amdgcn_tensor_load_to_lds(g0, g1, g2, g3, g4, 0);
#else
    __builtin_amdgcn_tensor_load_to_lds(g0, g1, g2, g3, 0);
#endif
}

// --------------------------- kernel 0: zero accumulators --------------------
__global__ void vol_zero(float* __restrict__ accum) {
    if (threadIdx.x < 4) accum[threadIdx.x] = 0.0f;
}

// --------------------------- kernel 1: residuals + u^2 ----------------------
__global__ void vol_resid(const float* __restrict__ r,
                          const float* __restrict__ pa0,
                          const float* __restrict__ pa1,
                          float*    __restrict__ u2full,   // [QLAG + n]
                          _Float16* __restrict__ u2h,      // [QLAG + n]
                          float*    __restrict__ accum,
                          int n)
{
    const float a0 = pa0[0], a1 = pa1[0];
    float s1 = 0.0f, s2 = 0.0f;
    for (int t = blockIdx.x * blockDim.x + threadIdx.x; t < n;
         t += gridDim.x * blockDim.x) {
        float u  = r[t + 1] - a0 - a1 * r[t];
        float u2 = u * u;
        u2full[QLAG + t] = u2;
        u2h[QLAG + t]    = (_Float16)(u2 * U2SCALE);
        s1 += u2;
        s2 += u2 * u2;
    }
    #pragma unroll
    for (int off = 16; off > 0; off >>= 1) {
        s1 += __shfl_xor(s1, off, 32);
        s2 += __shfl_xor(s2, off, 32);
    }
    if ((threadIdx.x & 31) == 0) {
        atomicAdd(&accum[0], s1);
        atomicAdd(&accum[1], s2);
    }
}

// --------------------------- kernel 2: parameter prep -----------------------
// wbT[k][l'] = beta[k] * softplus(raw_w[k][63-l'])   (beta folded into B)
// w0f[k]     = beta[k] * softplus(raw_w0[k]) * U2SCALE (bias, pre-scaled)
__global__ void vol_prep(const float* __restrict__ raw_beta0,
                         const float* __restrict__ raw_beta,
                         const float* __restrict__ raw_w0,
                         const float* __restrict__ raw_w,     // [256*64]
                         const float* __restrict__ accum,
                         float*    __restrict__ scal,
                         float*    __restrict__ u2full,
                         _Float16* __restrict__ u2h,
                         _Float16* __restrict__ wbT,           // [256*64]
                         float*    __restrict__ w0f,
                         int n)
{
    const int k = threadIdx.x;           // one block, 256 threads
    if (k < NKU) {
        const float bk = softplusf(raw_beta[k]);
        w0f[k] = bk * softplusf(raw_w0[k]) * U2SCALE;
        #pragma unroll 4
        for (int l = 0; l < QLAG; ++l)
            wbT[k * QLAG + l] =
                (_Float16)(bk * softplusf(raw_w[k * QLAG + (QLAG - 1 - l)]));
    }
    if (k == 0) {
        const float nn  = (float)n;
        const float var = (accum[1] - accum[0] * accum[0] / nn) / (nn - 1.0f);
        scal[0] = softplusf(raw_beta0[0]);
        const _Float16 vh = (_Float16)(var * U2SCALE);
        for (int l = 0; l < QLAG; ++l) { u2full[l] = var; u2h[l] = vh; }
    }
}

// --------------------------- kernel 3: WMMA main -----------------------------
struct SharedBlk {
    __align__(16) _Float16 wb[NKU * QLAG];   // 32 KB B-operand image (offset 0)
    float w0[NKU];
};

// Cross-lane reduce of one 16x16 D tile's relu-sums + NLL stats for 16 rows.
static __device__ __forceinline__ void reduce_stats(float* acc, int rbase,
                                                    int m, int hi,
                                                    const float* __restrict__ u2full,
                                                    float beta0,
                                                    float& lsum_log,
                                                    float& lsum_ratio)
{
    #pragma unroll
    for (int r = 0; r < 8; ++r) {
        float v = acc[r];
        v += __shfl_xor(v, 1, 32);
        v += __shfl_xor(v, 2, 32);
        v += __shfl_xor(v, 4, 32);
        v += __shfl_xor(v, 8, 32);
        acc[r] = v;
    }
    if (m == 0) {
        const int rb = rbase + (hi ? 8 : 0);
        #pragma unroll
        for (int r = 0; r < 8; ++r) {
            const float s2 = fmaf(acc[r], INV_U2SCALE, beta0 + 1e-8f);
            lsum_log   += __logf(s2);
            lsum_ratio += u2full[QLAG + rb + r] / s2;
        }
    }
}

__launch_bounds__(256)
__global__ void vol_wmma(const _Float16* __restrict__ u2h,    // [QLAG+n] scaled
                         const float*    __restrict__ u2full, // [QLAG+n]
                         const _Float16* __restrict__ wbT_g,  // [256*64]
                         const float*    __restrict__ w0_g,   // pre-scaled bias
                         const float*    __restrict__ scal,
                         float*          __restrict__ accum,
                         int n)
{
    __shared__ SharedBlk sm;

    const int tid = threadIdx.x;

    // ---- stage B into LDS: TDM from wave 0, plus benign-race ds copy -------
    if (tid < 32) {
        tdm_load_to_lds(wbT_g, (unsigned int)(unsigned long long)(&sm.wb[0]),
                        (NKU * QLAG * (int)sizeof(_Float16)) / 4);
        __builtin_amdgcn_s_wait_tensorcnt(0);
    }
    {
        const v8h* src = (const v8h*)wbT_g;
        v8h*       dst = (v8h*)sm.wb;
        #pragma unroll
        for (int i = 0; i < (NKU * QLAG / 8) / 256; ++i)
            dst[tid + i * 256] = src[tid + i * 256];
        sm.w0[tid] = w0_g[tid];
    }
    __syncthreads();

    const int lane        = tid & 31;
    const int wavesPerBlk = blockDim.x >> 5;
    const int wave        = blockIdx.x * wavesPerBlk + (tid >> 5);
    const int nWaves      = gridDim.x * wavesPerBlk;
    const int nTiles      = n >> 5;        // 500000 -> 15625 blocks of 32 rows

    const int m     = lane & 15;           // row-in-tile (A) / column (B,D)
    const int hi    = lane >> 4;           // lane half
    const int aoff  = hi ? 8 : 0;          // A K-offset (ISA layout)
    const int bslab = hi ? 1 : 0;          // B K-offset in v8h units

    // Per-lane bias values for all 16 column tiles (invariant across rows).
    float w0p[16];
    #pragma unroll
    for (int j = 0; j < 16; ++j) w0p[j] = sm.w0[(j << 4) + m];

    const float beta0 = scal[0];
    float lsum_log = 0.0f, lsum_ratio = 0.0f;

    for (int tile = wave; tile < nTiles; tile += nWaves) {
        const int t0 = tile << 5;          // 32 rows per iteration

        // A fragments (Hankel rows): A[m][k] = u2h[trow + m + k]
        v16h a0, a1, a2, a3;
        {
            const _Float16* p0 = u2h + (t0 + m + aoff);        // rows t0..+15
            const _Float16* p1 = p0 + 32;
            const _Float16* p2 = p0 + 16;                      // rows t0+16..+31
            const _Float16* p3 = p0 + 48;
            #pragma unroll
            for (int j = 0; j < 8; ++j) {
                a0[j] = p0[j];  a0[8 + j] = p0[16 + j];
                a1[j] = p1[j];  a1[8 + j] = p1[16 + j];
                a2[j] = p2[j];  a2[8 + j] = p2[16 + j];
                a3[j] = p3[j];  a3[8 + j] = p3[16 + j];
            }
        }
        __builtin_prefetch(u2h + t0 + (nWaves << 5), 0, 3);   // global_prefetch

        float acc0[8], acc1[8];
        #pragma unroll
        for (int r = 0; r < 8; ++r) { acc0[r] = 0.0f; acc1[r] = 0.0f; }

        #pragma unroll 2
        for (int j = 0; j < 16; ++j) {
            const int ncol = (j << 4) + m;

            // B fragments shared by both row tiles (4 WMMAs per fetch pair)
            const v8h* row = (const v8h*)(sm.wb + ncol * QLAG);
            v16h b0 = cat8(row[bslab],     row[bslab + 2]);   // K 0..31
            v16h b1 = cat8(row[bslab + 4], row[bslab + 6]);   // K 32..63

            v8f c0, c1;
            #pragma unroll
            for (int r = 0; r < 8; ++r) { c0[r] = w0p[j]; c1[r] = w0p[j]; }

            v8f d0 = __builtin_amdgcn_wmma_f32_16x16x32_f16(
                        false, a0, false, b0, (short)0, c0, false, false);
            v8f d1 = __builtin_amdgcn_wmma_f32_16x16x32_f16(
                        false, a2, false, b0, (short)0, c1, false, false);
            d0     = __builtin_amdgcn_wmma_f32_16x16x32_f16(
                        false, a1, false, b1, (short)0, d0, false, false);
            d1     = __builtin_amdgcn_wmma_f32_16x16x32_f16(
                        false, a3, false, b1, (short)0, d1, false, false);

            #pragma unroll
            for (int r = 0; r < 8; ++r) {
                acc0[r] += reluf(d0[r]);          // beta folded; 1/S applied in reduce_stats
                acc1[r] += reluf(d1[r]);
            }
        }

        reduce_stats(acc0, t0,      m, hi, u2full, beta0, lsum_log, lsum_ratio);
        reduce_stats(acc1, t0 + 16, m, hi, u2full, beta0, lsum_log, lsum_ratio);
    }

    lsum_log   += __shfl_xor(lsum_log,   16, 32);
    lsum_ratio += __shfl_xor(lsum_ratio, 16, 32);
    if (lane == 0) {
        atomicAdd(&accum[2], lsum_log);
        atomicAdd(&accum[3], lsum_ratio);
    }
}

// --------------------------- kernel 4: finalize ------------------------------
__global__ void vol_final(const float* __restrict__ accum,
                          float* __restrict__ out, int n)
{
    out[0] = 0.5f * (float)n * 1.8378770664093453f   // log(2*pi)
           + 0.5f * accum[2] + 0.5f * accum[3];
}

// ---------------------------------------------------------------------------
extern "C" void kernel_launch(void* const* d_in, const int* in_sizes, int n_in,
                              void* d_out, int out_size, void* d_ws, size_t ws_size,
                              hipStream_t stream)
{
    const float* r         = (const float*)d_in[0];
    const float* a0        = (const float*)d_in[1];
    const float* a1        = (const float*)d_in[2];
    const float* raw_beta0 = (const float*)d_in[3];
    const float* raw_beta  = (const float*)d_in[4];
    const float* raw_w0    = (const float*)d_in[5];
    const float* raw_w     = (const float*)d_in[6];

    const int T = in_sizes[0];
    const int n = T - 1;                       // 500000 (multiple of 32)

    char*  ws  = (char*)d_ws;
    size_t off = 0;
    auto alloc = [&](size_t bytes) -> char* {
        char* p = ws + off;
        off = (off + bytes + 15) & ~(size_t)15;
        return p;
    };
    float*    accum  = (float*)   alloc(4 * sizeof(float));
    float*    scal   = (float*)   alloc(4 * sizeof(float));
    float*    u2full = (float*)   alloc((size_t)(QLAG + n) * sizeof(float));
    _Float16* u2h    = (_Float16*)alloc((size_t)(QLAG + n) * sizeof(_Float16));
    _Float16* wbT    = (_Float16*)alloc((size_t)NKU * QLAG * sizeof(_Float16));
    float*    w0f    = (float*)   alloc(NKU * sizeof(float));
    (void)ws_size;

    vol_zero<<<1, 32, 0, stream>>>(accum);
    vol_resid<<<1024, 256, 0, stream>>>(r, a0, a1, u2full, u2h, accum, n);
    vol_prep<<<1, 256, 0, stream>>>(raw_beta0, raw_beta, raw_w0, raw_w, accum,
                                    scal, u2full, u2h, wbT, w0f, n);
    vol_wmma<<<512, 256, 0, stream>>>(u2h, u2full, wbT, w0f, scal, accum, n);
    vol_final<<<1, 1, 0, stream>>>(accum, (float*)d_out, n);
}
